// GCPNetModel_32083405701172
// MI455X (gfx1250) — matrix operand order repaired and checked by reading, not compiled, not run on hardware
//
// CDNA5 / gfx1250 implementation of the GCPNet reference.
//
// Strategy: all W_s matmuls (M up to 800k items, K 24..448, N 32/128) run as
// bf16 WMMA GEMMs (v_wmma_f32_16x16x32_bf16, f32 accumulate). Per-GCP
// pipeline: pre (gather + vh/vn/frame-scalarization -> bf16 feats), WMMA GEMM
// (+bias), post (SiLU, sigmoid gate, W_vo, optional atomic segment-sum
// scatter). LayerNorms use wave32 shuffle reductions.
//
// GEMM kernel: each block owns 64 rows (4 M-subtiles); per K-step one B
// fragment feeds 4 WMMAs (4x B reuse). KPAD is a template constant so all
// A/B accesses fold into 24-bit immediate offsets off a single base register
// pair each -> compiler software-pipelines with rotating fragment registers
// (s_wait_loadcnt > 0) instead of stalling per-wmma.
//
// Workspace budget ~1.75 GB in d_ws (checked vs ws_size). All launches on
// `stream`; only hipMemsetAsync + kernel launches (graph-capture safe).
//
// Param pointers d_in[7..] assume JAX pytree flattening of the params dict
// (dict keys sorted): embed_edge{W_g,W_s,W_vh,W_vo,b_s}, embed_node{...},
// layer0..4{msg{...},upd{...}}, proj{W_s,W_vh,b_s}.

#include <hip/hip_runtime.h>
#include <math.h>

// ---------------------------------------------------------------- types ----
typedef __attribute__((ext_vector_type(16))) __bf16 v16bf;
typedef __attribute__((ext_vector_type(8)))  __bf16 v8bf;
typedef __attribute__((ext_vector_type(8)))  float  v8f;

union AFrag { struct { v8bf lo, hi; } p; v16bf v; };

__device__ __forceinline__ unsigned short f2bf(float f) {
  union { float f; unsigned u; } v; v.f = f;
  unsigned r = v.u + 0x7FFFu + ((v.u >> 16) & 1u);   // round-to-nearest-even
  return (unsigned short)(r >> 16);
}

#define GCP_EPS 1e-8f

// ------------------------------------------------- geometry / reductions ----
__global__ void k_cent_acc(const float* __restrict__ pos,
                           const int* __restrict__ batch,
                           float* __restrict__ cent, int n) {
  int i = blockIdx.x * blockDim.x + threadIdx.x;
  if (i >= n) return;
  int b = batch[i];
  atomicAdd(&cent[b * 4 + 0], pos[i * 3 + 0]);
  atomicAdd(&cent[b * 4 + 1], pos[i * 3 + 1]);
  atomicAdd(&cent[b * 4 + 2], pos[i * 3 + 2]);
  atomicAdd(&cent[b * 4 + 3], 1.0f);
}

__global__ void k_center(const float* __restrict__ pos,
                         const int* __restrict__ batch,
                         const float* __restrict__ cent,
                         float* __restrict__ pos_c, int n) {
  int i = blockIdx.x * blockDim.x + threadIdx.x;
  if (i >= n) return;
  int b = batch[i];
  float inv = 1.0f / fmaxf(cent[b * 4 + 3], 1.0f);
  pos_c[i * 3 + 0] = pos[i * 3 + 0] - cent[b * 4 + 0] * inv;
  pos_c[i * 3 + 1] = pos[i * 3 + 1] - cent[b * 4 + 1] * inv;
  pos_c[i * 3 + 2] = pos[i * 3 + 2] - cent[b * 4 + 2] * inv;
}

__global__ void k_frames(const float* __restrict__ pos_c,
                         const int* __restrict__ row,
                         const int* __restrict__ col,
                         float* __restrict__ fr, int E) {
  int e = blockIdx.x * blockDim.x + threadIdx.x;
  if (e >= E) return;
  int r = row[e], c = col[e];
  float p0 = pos_c[(size_t)r * 3 + 0], p1 = pos_c[(size_t)r * 3 + 1], p2 = pos_c[(size_t)r * 3 + 2];
  float q0 = pos_c[(size_t)c * 3 + 0], q1 = pos_c[(size_t)c * 3 + 1], q2 = pos_c[(size_t)c * 3 + 2];
  float d0 = p0 - q0, d1 = p1 - q1, d2 = p2 - q2;
  float ia = rsqrtf(d0 * d0 + d1 * d1 + d2 * d2 + GCP_EPS);
  float a0 = d0 * ia, a1 = d1 * ia, a2 = d2 * ia;
  float x0 = p1 * q2 - p2 * q1, x1 = p2 * q0 - p0 * q2, x2 = p0 * q1 - p1 * q0;
  float ib = rsqrtf(x0 * x0 + x1 * x1 + x2 * x2 + GCP_EPS);
  float b0 = x0 * ib, b1 = x1 * ib, b2 = x2 * ib;
  float c0 = a1 * b2 - a2 * b1, c1 = a2 * b0 - a0 * b2, c2 = a0 * b1 - a1 * b0;
  float* f = fr + (size_t)e * 9;
  f[0] = a0; f[1] = a1; f[2] = a2;
  f[3] = b0; f[4] = b1; f[5] = b2;
  f[6] = c0; f[7] = c1; f[8] = c2;
}

// ------------------------------------------ weight prep: f32 -> B-fragment ----
// B tile layout for v_wmma_f32_16x16x32_bf16 (ISA 7.12.2): tile (kt,nt),
// lane elem j holds W[kt*32 + (lane>>4)*8 + j + (j>=8 ? 8 : 0)][nt*16 + (lane&15)].
__global__ void k_prep_w(const float* __restrict__ W, unsigned short* __restrict__ out,
                         int K, int Kpad, int Nd) {
  int idx = blockIdx.x * blockDim.x + threadIdx.x;
  int ntiles = Nd >> 4;
  int total = (Kpad >> 5) * ntiles * 32;
  if (idx >= total) return;
  int lane = idx & 31;
  int tile = idx >> 5;
  int nt = tile % ntiles;
  int kt = tile / ntiles;
  int n = nt * 16 + (lane & 15);
  int kb = kt * 32 + ((lane >> 4) << 3);
  unsigned short* dst = out + ((size_t)(kt * ntiles + nt) * 32 + lane) * 16;
#pragma unroll
  for (int j = 0; j < 16; ++j) {
    int k = kb + j + (j >= 8 ? 8 : 0);
    dst[j] = (k < K) ? f2bf(W[(size_t)k * Nd + n]) : (unsigned short)0;
  }
}

// ----------------------------------------------------------- GCP "pre" ----
// VAR: 0=embed_node 1=embed_edge 2=msg 3=upd 4=proj
struct PreArgs {
  const float* s0; const float* s1;          // scalar sources
  const float* v0; const float* v1;          // vector sources
  const float* frames;
  const int* row; const int* col;
  const float* Wvh;                          // [dv_in][hv]
  unsigned short* feats;                     // [M][KPAD] bf16
  float* vh_out;                             // [M][hv][3]
};

template <int VAR>
__global__ __launch_bounds__(256) void k_gcp_pre(PreArgs a, int M) {
  constexpr int DSIN = (VAR == 0) ? 64 : (VAR == 1) ? 8 : (VAR == 2) ? 288 : (VAR == 3) ? 256 : 128;
  constexpr int DVIN = (VAR == 0) ? 2 : (VAR == 1) ? 1 : (VAR == 2) ? 36 : (VAR == 3) ? 32 : 16;
  constexpr int HV   = (VAR == 0) ? 16 : (VAR == 1) ? 4 : (VAR == 2) ? 36 : (VAR == 3) ? 32 : 16;
  constexpr int FR   = (VAR == 1 || VAR == 2) ? 1 : 0;
  constexpr int SIN  = DSIN + HV + (FR ? 3 * HV : 0);
  constexpr int KPAD = (SIN + 31) & ~31;

  __shared__ float sW[DVIN * HV];
  __shared__ float sv[16 * DVIN * 3];
  __shared__ float sFr[FR ? 16 * 9 : 1];
  __shared__ int   sR[16], sC[16];

  const int tid = threadIdx.x;
  const size_t tile = blockIdx.x;

  if (VAR == 2) {
    if (tid < 16) {
      sR[tid] = a.row[tile * 16 + tid];
      sC[tid] = a.col[tile * 16 + tid];
    }
  }
  for (int i = tid; i < DVIN * HV; i += 256) sW[i] = a.Wvh[i];
  if constexpr (FR) {
    for (int i = tid; i < 16 * 9; i += 256) sFr[i] = a.frames[tile * 144 + i];
  }
  __syncthreads();

  // gather vector inputs into LDS: sv[item][cc][dim]
  for (int i = tid; i < 16 * DVIN * 3; i += 256) {
    int it = i / (DVIN * 3);
    int rem = i - it * (DVIN * 3);
    int cc = rem / 3, d = rem - cc * 3;
    size_t m = tile * 16 + it;
    float val;
    if (VAR == 0)      val = a.v0[(m * 2 + cc) * 3 + d];
    else if (VAR == 1) val = a.v0[m * 3 + d];
    else if (VAR == 2) {
      if (cc < 16)      val = a.v0[((size_t)sR[it] * 16 + cc) * 3 + d];
      else if (cc < 32) val = a.v0[((size_t)sC[it] * 16 + (cc - 16)) * 3 + d];
      else              val = a.v1[(m * 4 + (cc - 32)) * 3 + d];
    } else if (VAR == 3) {
      if (cc < 16) val = a.v0[(m * 16 + cc) * 3 + d];
      else         val = a.v1[(m * 16 + (cc - 16)) * 3 + d];
    } else             val = a.v0[(m * 16 + cc) * 3 + d];
    sv[i] = val;
  }
  __syncthreads();

  // scalar slice of feats row
  for (int i = tid; i < 16 * DSIN; i += 256) {
    int it = i / DSIN, cc = i - it * DSIN;
    size_t m = tile * 16 + it;
    float val;
    if (VAR == 0)      val = a.s0[m * 64 + cc];
    else if (VAR == 1) val = a.s0[m * 8 + cc];
    else if (VAR == 2) {
      if (cc < 128)      val = a.s0[(size_t)sR[it] * 128 + cc];
      else if (cc < 256) val = a.s0[(size_t)sC[it] * 128 + (cc - 128)];
      else               val = a.s1[m * 32 + (cc - 256)];
    } else if (VAR == 3) {
      if (cc < 128) val = a.s0[m * 128 + cc];
      else          val = a.s1[m * 128 + (cc - 128)];
    } else             val = a.s0[m * 128 + cc];
    a.feats[m * KPAD + cc] = f2bf(val);
  }
  if constexpr (KPAD > SIN) {
    for (int i = tid; i < 16 * (KPAD - SIN); i += 256) {
      int it = i / (KPAD - SIN), cc = SIN + i % (KPAD - SIN);
      a.feats[(tile * 16 + it) * (size_t)KPAD + cc] = 0;
    }
  }

  // vh, |vh|, frame scalarization: 16 threads per item
  const int it = tid >> 4, t = tid & 15;
  const size_t m = tile * 16 + it;
  const float* svi = &sv[it * DVIN * 3];
  for (int h = t; h < HV; h += 16) {
    float x = 0.f, y = 0.f, z = 0.f;
    for (int cc = 0; cc < DVIN; ++cc) {
      float w = sW[cc * HV + h];
      x += w * svi[cc * 3 + 0];
      y += w * svi[cc * 3 + 1];
      z += w * svi[cc * 3 + 2];
    }
    float* vo = &a.vh_out[(m * HV + h) * 3];
    vo[0] = x; vo[1] = y; vo[2] = z;
    a.feats[m * KPAD + DSIN + h] = f2bf(sqrtf(x * x + y * y + z * z + GCP_EPS));
    if constexpr (FR) {
#pragma unroll
      for (int rr = 0; rr < 3; ++rr) {
        float d0 = sFr[it * 9 + rr * 3 + 0];
        float d1 = sFr[it * 9 + rr * 3 + 1];
        float d2 = sFr[it * 9 + rr * 3 + 2];
        a.feats[m * KPAD + DSIN + HV + h * 3 + rr] = f2bf(x * d0 + y * d1 + z * d2);
      }
    }
  }
}

// ----------------------------------------------------------- WMMA GEMM ----
// spre[M][Nd] = A[M][KPAD](bf16) x B(frag-layout bf16) + bias.
// Each block owns 64 rows = 4 M-subtiles; wave w owns N cols [16w,16w+16).
// Per K-step: one B fragment (2 x b128) drives 4 WMMAs (4x B reuse).
// KPAD/strides are compile-time constants: every load is base + immediate
// offset (A max ~44KB, B max ~104KB, both << 24-bit IOFFSET), so the
// compiler can pipeline loads ahead of the WMMAs with rotating registers.
template <int NT, int KPAD>
__global__ __launch_bounds__(NT * 32) void k_gemm(
    const unsigned short* __restrict__ A, const unsigned short* __restrict__ B,
    const float* __restrict__ bias, float* __restrict__ spre,
    int Nd, int M) {
  const int w = threadIdx.x >> 5;
  const int lane = threadIdx.x & 31;
  const int half = lane >> 4;
  const int lrow = lane & 15;
  const size_t tile0 = (size_t)blockIdx.x * 4;          // first 16-row subtile

  constexpr int ASTEP = 16 * KPAD;                      // elems per subtile
  constexpr int BSTEP = NT * 32 * 16;                   // elems per K-step
  const unsigned short* abase = A + (tile0 * 16 + lrow) * (size_t)KPAD + half * 8;
  const unsigned short* bbase = B + ((size_t)w * 32 + lane) * 16;

  v8f acc0{}, acc1{}, acc2{}, acc3{};
#pragma unroll
  for (int kt = 0; kt < KPAD; kt += 32) {
    const int bo = (kt / 32) * BSTEP;
    v16bf bf = *(const v16bf*)(bbase + bo);
    AFrag a0, a1, a2, a3;
    a0.p.lo = *(const v8bf*)(abase + kt);                  // K = kt+half*8 .. +7
    a0.p.hi = *(const v8bf*)(abase + kt + 16);             // K = kt+16+half*8 .. +7
    a1.p.lo = *(const v8bf*)(abase + ASTEP + kt);
    a1.p.hi = *(const v8bf*)(abase + ASTEP + kt + 16);
    a2.p.lo = *(const v8bf*)(abase + 2 * ASTEP + kt);
    a2.p.hi = *(const v8bf*)(abase + 2 * ASTEP + kt + 16);
    a3.p.lo = *(const v8bf*)(abase + 3 * ASTEP + kt);
    a3.p.hi = *(const v8bf*)(abase + 3 * ASTEP + kt + 16);
    acc0 = __builtin_amdgcn_wmma_f32_16x16x32_bf16(false, a0.v, false, bf, (short)0, acc0, false, false);
    acc1 = __builtin_amdgcn_wmma_f32_16x16x32_bf16(false, a1.v, false, bf, (short)0, acc1, false, false);
    acc2 = __builtin_amdgcn_wmma_f32_16x16x32_bf16(false, a2.v, false, bf, (short)0, acc2, false, false);
    acc3 = __builtin_amdgcn_wmma_f32_16x16x32_bf16(false, a3.v, false, bf, (short)0, acc3, false, false);
  }
  const int n = w * 16 + lrow;
  const float bv = bias ? bias[n] : 0.f;
  const int mtiles = M >> 4;
  v8f accs[4] = {acc0, acc1, acc2, acc3};
#pragma unroll
  for (int t = 0; t < 4; ++t) {
    if (tile0 + t < (size_t)mtiles) {
      const size_t mrow = (tile0 + t) * 16 + half * 8;  // C: VGPR r -> M = half*8+r
#pragma unroll
      for (int r = 0; r < 8; ++r)
        spre[(mrow + r) * (size_t)Nd + n] = accs[t][r] + bv;
    }
  }
}

// ----------------------------------------------------------- GCP "post" ----
// SiLU + sigmoid gate (spre*W_g) + vh*W_vo; direct store or atomic scatter.
template <int DSO, int HV, int DVO, int SCATTER>
__global__ __launch_bounds__(256) void k_gcp_post(
    const float* __restrict__ spre, const float* __restrict__ vh,
    const float* __restrict__ Wg, const float* __restrict__ Wvo,
    float* __restrict__ sdst, float* __restrict__ vdst,
    const int* __restrict__ sidx, int M) {
  const int it = threadIdx.x >> 5, lane = threadIdx.x & 31;
  const size_t m = (size_t)blockIdx.x * 8 + it;
  if (m >= (size_t)M) return;
  const float* sp = spre + m * DSO;
  const size_t drow = SCATTER ? (size_t)sidx[m] : m;
  for (int c = lane; c < DSO; c += 32) {
    float x = sp[c];
    float s = x * (1.f / (1.f + __expf(-x)));   // silu
    if (SCATTER) atomicAdd(&sdst[drow * DSO + c], s);
    else sdst[m * DSO + c] = s;
  }
  if constexpr (DVO > 0) {
    if (lane < DVO) {
      const int o = lane;
      float g = 0.f;
      for (int h = 0; h < DSO; ++h) g += sp[h] * Wg[h * DVO + o];
      g = 1.f / (1.f + __expf(-g));             // sigmoid gate
      const float* vhm = vh + m * (size_t)HV * 3;
      float a0 = 0.f, a1 = 0.f, a2 = 0.f;
      for (int h = 0; h < HV; ++h) {
        float wv = Wvo[h * DVO + o];
        a0 += wv * vhm[h * 3 + 0];
        a1 += wv * vhm[h * 3 + 1];
        a2 += wv * vhm[h * 3 + 2];
      }
      float* vd = vdst + drow * (size_t)DVO * 3 + o * 3;
      if (SCATTER) {
        atomicAdd(vd + 0, a0 * g);
        atomicAdd(vd + 1, a1 * g);
        atomicAdd(vd + 2, a2 * g);
      } else {
        vd[0] = a0 * g; vd[1] = a1 * g; vd[2] = a2 * g;
      }
    }
  }
}

// ------------------------------------------- residual + sv_layernorm ----
__global__ __launch_bounds__(256) void k_res_ln(
    float* __restrict__ h, float* __restrict__ chi,
    const float* __restrict__ us, const float* __restrict__ uv,
    int M, int residual) {
  const int it = threadIdx.x >> 5, lane = threadIdx.x & 31;
  const size_t m = (size_t)blockIdx.x * 8 + it;
  if (m >= (size_t)M) return;
  float x[4];
  float sum = 0.f, sq = 0.f;
#pragma unroll
  for (int k = 0; k < 4; ++k) {
    int c = lane + 32 * k;
    float v = h[m * 128 + c];
    if (residual) v += us[m * 128 + c];
    x[k] = v; sum += v; sq += v * v;
  }
  for (int msk = 16; msk >= 1; msk >>= 1) {
    sum += __shfl_xor(sum, msk, 32);
    sq  += __shfl_xor(sq,  msk, 32);
  }
  const float mu = sum * (1.f / 128.f);
  const float var = sq * (1.f / 128.f) - mu * mu;
  const float inv = rsqrtf(var + GCP_EPS);
#pragma unroll
  for (int k = 0; k < 4; ++k) {
    int c = lane + 32 * k;
    h[m * 128 + c] = (x[k] - mu) * inv;
  }
  float v3[3] = {0.f, 0.f, 0.f};
  float vsq = 0.f;
  if (lane < 16) {
#pragma unroll
    for (int d = 0; d < 3; ++d) {
      float v = chi[(m * 16 + lane) * 3 + d];
      if (residual) v += uv[(m * 16 + lane) * 3 + d];
      v3[d] = v; vsq += v * v;
    }
  }
  for (int msk = 16; msk >= 1; msk >>= 1) vsq += __shfl_xor(vsq, msk, 32);
  const float inv2 = rsqrtf(vsq * (1.f / 16.f) + GCP_EPS);
  if (lane < 16) {
#pragma unroll
    for (int d = 0; d < 3; ++d)
      chi[(m * 16 + lane) * 3 + d] = v3[d] * inv2;
  }
}

// ------------------------------------------------------------- pooling ----
__global__ void k_pool(const float* __restrict__ node_emb,
                       const int* __restrict__ batch,
                       float* __restrict__ gout, int n) {
  int idx = blockIdx.x * blockDim.x + threadIdx.x;
  if (idx >= n * 128) return;
  int i = idx >> 7, c = idx & 127;
  atomicAdd(&gout[(size_t)batch[i] * 128 + c], node_emb[idx]);
}

// ================================================================ host ====
extern "C" void kernel_launch(void* const* d_in, const int* in_sizes, int n_in,
                              void* d_out, int out_size, void* d_ws, size_t ws_size,
                              hipStream_t stream) {
  const int N = in_sizes[0] / 3;
  const int E = in_sizes[5] / 2;
  const int G = 32;

  const float* pos = (const float*)d_in[0];
  const float* x   = (const float*)d_in[1];
  const float* xv  = (const float*)d_in[2];
  const float* ea  = (const float*)d_in[3];
  const float* ev  = (const float*)d_in[4];
  const int* ei    = (const int*)d_in[5];
  const int* batch = (const int*)d_in[6];
  const int* row = ei;
  const int* col = ei + E;

  struct GW { const float *Wg, *Ws, *Wvh, *Wvo, *bs; };
  int p = 7;
  auto take5 = [&]() {
    GW g;
    g.Wg  = (const float*)d_in[p++];
    g.Ws  = (const float*)d_in[p++];
    g.Wvh = (const float*)d_in[p++];
    g.Wvo = (const float*)d_in[p++];
    g.bs  = (const float*)d_in[p++];
    return g;
  };
  GW ee = take5();           // embed_edge
  GW en = take5();           // embed_node
  GW msg[5], upd[5];
  for (int l = 0; l < 5; ++l) { msg[l] = take5(); upd[l] = take5(); }
  const float* pj_Ws  = (const float*)d_in[p++];
  const float* pj_Wvh = (const float*)d_in[p++];
  const float* pj_bs  = (const float*)d_in[p++];
  (void)n_in; (void)out_size;

  // -------- workspace carve-up (256B aligned) --------
  size_t off = 0;
  auto alloc = [&](size_t bytes) {
    size_t o = (off + 255) & ~(size_t)255;
    off = o + bytes;
    return (void*)((char*)d_ws + o);
  };
  float* cent   = (float*)alloc((size_t)G * 4 * 4);
  float* pos_c  = (float*)alloc((size_t)N * 3 * 4);
  float* fr     = (float*)alloc((size_t)E * 9 * 4);
  float* h      = (float*)alloc((size_t)N * 128 * 4);
  float* chi    = (float*)alloc((size_t)N * 48 * 4);
  float* h2     = (float*)alloc((size_t)N * 128 * 4);
  float* chi2   = (float*)alloc((size_t)N * 48 * 4);
  float* ebuf   = (float*)alloc((size_t)E * 32 * 4);
  float* xibuf  = (float*)alloc((size_t)E * 12 * 4);
  float* agg_s  = (float*)alloc((size_t)N * 128 * 4);
  float* agg_v  = (float*)alloc((size_t)N * 48 * 4);
  unsigned short* feats = (unsigned short*)alloc((size_t)E * 448 * 2);
  float* vhbuf  = (float*)alloc((size_t)E * 36 * 3 * 4);
  float* spre   = (float*)alloc((size_t)E * 128 * 4);
  unsigned short* wN = (unsigned short*)alloc((size_t)96 * 128 * 2);
  unsigned short* wE = (unsigned short*)alloc((size_t)32 * 32 * 2);
  unsigned short *wM[5], *wU[5];
  for (int l = 0; l < 5; ++l) {
    wM[l] = (unsigned short*)alloc((size_t)448 * 128 * 2);
    wU[l] = (unsigned short*)alloc((size_t)288 * 128 * 2);
  }
  unsigned short* wP = (unsigned short*)alloc((size_t)160 * 128 * 2);
  if (off > ws_size) return;   // workspace too small; nothing safe to do

  const int ntile_N = (N / 16 + 3) / 4;   // 4 subtiles per GEMM block
  const int ntile_E = (E / 16 + 3) / 4;

  // -------- geometry --------
  hipMemsetAsync(cent, 0, (size_t)G * 4 * 4, stream);
  k_cent_acc<<<(N + 255) / 256, 256, 0, stream>>>(pos, batch, cent, N);
  k_center<<<(N + 255) / 256, 256, 0, stream>>>(pos, batch, cent, pos_c, N);
  k_frames<<<(E + 255) / 256, 256, 0, stream>>>(pos_c, row, col, fr, E);

  // -------- weight prep (f32 -> bf16 WMMA B-fragments) --------
  auto prep = [&](const float* W, unsigned short* dst, int K, int Kpad, int Nd) {
    int total = (Kpad / 32) * (Nd / 16) * 32;
    k_prep_w<<<(total + 255) / 256, 256, 0, stream>>>(W, dst, K, Kpad, Nd);
  };
  prep(en.Ws, wN, 80, 96, 128);
  prep(ee.Ws, wE, 24, 32, 32);
  for (int l = 0; l < 5; ++l) {
    prep(msg[l].Ws, wM[l], 432, 448, 128);
    prep(upd[l].Ws, wU[l], 288, 288, 128);
  }
  prep(pj_Ws, wP, 144, 160, 128);

  // -------- node embedding GCP --------
  {
    PreArgs a{};
    a.s0 = x; a.v0 = xv; a.Wvh = en.Wvh; a.feats = feats; a.vh_out = vhbuf;
    k_gcp_pre<0><<<N / 16, 256, 0, stream>>>(a, N);
    k_gemm<8, 96><<<ntile_N, 256, 0, stream>>>(feats, wN, en.bs, spre, 128, N);
    k_gcp_post<128, 16, 16, 0><<<(N + 7) / 8, 256, 0, stream>>>(
        spre, vhbuf, en.Wg, en.Wvo, h, chi, nullptr, N);
  }
  // -------- edge embedding GCP (with frames) --------
  {
    PreArgs a{};
    a.s0 = ea; a.v0 = ev; a.frames = fr; a.Wvh = ee.Wvh;
    a.feats = feats; a.vh_out = vhbuf;
    k_gcp_pre<1><<<E / 16, 256, 0, stream>>>(a, E);
    k_gemm<2, 32><<<ntile_E, 64, 0, stream>>>(feats, wE, ee.bs, spre, 32, E);
    k_gcp_post<32, 4, 4, 0><<<(E + 7) / 8, 256, 0, stream>>>(
        spre, vhbuf, ee.Wg, ee.Wvo, ebuf, xibuf, nullptr, E);
  }

  // -------- interaction layers --------
  for (int l = 0; l < 5; ++l) {
    hipMemsetAsync(agg_s, 0, (size_t)N * 128 * 4, stream);
    hipMemsetAsync(agg_v, 0, (size_t)N * 48 * 4, stream);
    // message GCP over edges (gather + frames), scatter-add to nodes
    {
      PreArgs a{};
      a.s0 = h; a.s1 = ebuf; a.v0 = chi; a.v1 = xibuf;
      a.frames = fr; a.row = row; a.col = col; a.Wvh = msg[l].Wvh;
      a.feats = feats; a.vh_out = vhbuf;
      k_gcp_pre<2><<<E / 16, 256, 0, stream>>>(a, E);
      k_gemm<8, 448><<<ntile_E, 256, 0, stream>>>(feats, wM[l], msg[l].bs, spre, 128, E);
      k_gcp_post<128, 36, 16, 1><<<(E + 7) / 8, 256, 0, stream>>>(
          spre, vhbuf, msg[l].Wg, msg[l].Wvo, agg_s, agg_v, row, E);
    }
    // update GCP over nodes
    {
      PreArgs a{};
      a.s0 = h; a.s1 = agg_s; a.v0 = chi; a.v1 = agg_v; a.Wvh = upd[l].Wvh;
      a.feats = feats; a.vh_out = vhbuf;
      k_gcp_pre<3><<<N / 16, 256, 0, stream>>>(a, N);
      k_gemm<8, 288><<<ntile_N, 256, 0, stream>>>(feats, wU[l], upd[l].bs, spre, 128, N);
      k_gcp_post<128, 32, 16, 0><<<(N + 7) / 8, 256, 0, stream>>>(
          spre, vhbuf, upd[l].Wg, upd[l].Wvo, h2, chi2, nullptr, N);
    }
    // h, chi = LN(h + u_s, chi + u_v)
    k_res_ln<<<(N + 7) / 8, 256, 0, stream>>>(h, chi, h2, chi2, N, 1);
  }

  // -------- final LN + invariant projection + pooling --------
  k_res_ln<<<(N + 7) / 8, 256, 0, stream>>>(h, chi, nullptr, nullptr, N, 0);
  {
    PreArgs a{};
    a.s0 = h; a.v0 = chi; a.Wvh = pj_Wvh; a.feats = feats; a.vh_out = vhbuf;
    k_gcp_pre<4><<<N / 16, 256, 0, stream>>>(a, N);
    k_gemm<8, 160><<<ntile_N, 256, 0, stream>>>(feats, wP, pj_bs, spre, 128, N);
    k_gcp_post<128, 16, 0, 0><<<(N + 7) / 8, 256, 0, stream>>>(
        spre, nullptr, nullptr, nullptr, (float*)d_out, nullptr, nullptr, N);
  }
  float* gout = (float*)d_out + (size_t)N * 128;
  hipMemsetAsync(gout, 0, (size_t)G * 128 * 4, stream);
  k_pool<<<((size_t)N * 128 + 255) / 256, 256, 0, stream>>>(
      (const float*)d_out, batch, gout, N);
}
